// EMLTree_30803505447173
// MI455X (gfx1250) — matrix operand order repaired and checked by reading, not compiled, not run on hardware
//
#include <hip/hip_runtime.h>
#include <hip/hip_fp16.h>

// ---------------------------------------------------------------------------
// EML soft-tree:  out[b] = log sum_l exp( bias[l] + sum_k X[b,k]*W[l,k+1] )
// W = softmax(leaf_logits).  GEMM 8192x1024x512 in f16 WMMA (f32 acc),
// exp + leaf logsumexp fused into the GEMM epilogue.
//
// GEMM block: 128 rows x 128 cols, 8 waves (wave32) as 2(M) x 4(N);
// wave tile 64x32 = 4 m-frags x 2 n-frags = 8 WMMA accumulators.
// A tile (128 x 512 f16 = 128KB, contiguous because pre-swizzled) is staged
// into LDS by the Tensor Data Mover (one tensor_load_to_lds + s_wait_tensorcnt)
// with an async-load fallback; B streams from L2 (1MB resident).
// ---------------------------------------------------------------------------

typedef __attribute__((ext_vector_type(16))) _Float16 v16h;
typedef __attribute__((ext_vector_type(8)))  float    v8f;
typedef __attribute__((ext_vector_type(4)))  unsigned int u32x4;
typedef __attribute__((ext_vector_type(8)))  int          i32x8;
typedef __attribute__((ext_vector_type(4)))  int          i32x4;

#define BATCH   8192
#define NF      512          // GEMM K
#define NL      1024         // leaves (GEMM N)
#define ATOMS   513
#define KT      16           // K tiles of 32
#define BR      128          // rows per block   (8 m-tiles)
#define BC      128          // cols per block   (8 n-tiles)
#define CB      (NL / BC)    // 8 column blocks

// ---------------------------------------------------------------------------
// P1: row softmax of leaf_logits; emit bias[l]=w[l,0] and B fragments in the
// WMMA 16-bit B 32x16 per-lane layout (lanes 0-15: K 0..7,16..23, lanes 16-31:
// K 8..15,24..31, col = lane&15).  One wave per leaf.
// ---------------------------------------------------------------------------
__global__ __launch_bounds__(256) void eml_prep_weights(
    const float* __restrict__ ll, _Float16* __restrict__ Bswz,
    float* __restrict__ bias) {
  const int wave = threadIdx.x >> 5;
  const int lane = threadIdx.x & 31;
  const int l = blockIdx.x * 8 + wave;
  const float* row = ll + l * ATOMS;

  float mx = -1e30f;
  for (int a = lane; a < ATOMS; a += 32) mx = fmaxf(mx, row[a]);
  #pragma unroll
  for (int m = 16; m >= 1; m >>= 1) mx = fmaxf(mx, __shfl_xor(mx, m, 32));

  float s = 0.0f;
  for (int a = lane; a < ATOMS; a += 32) s += __expf(row[a] - mx);
  #pragma unroll
  for (int m = 16; m >= 1; m >>= 1) s += __shfl_xor(s, m, 32);
  const float inv = 1.0f / s;

  if (lane == 0) bias[l] = __expf(row[0] - mx) * inv;

  const int nt = l >> 4;
  const int nc = l & 15;
  for (int k = lane; k < NF; k += 32) {
    float w = __expf(row[k + 1] - mx) * inv;
    const int kt = k >> 5;
    const int ko = k & 31;
    const int bl = nc + ((ko & 8) ? 16 : 0);
    const int j  = (ko & 7) | ((ko & 16) >> 1);
    Bswz[(((size_t)(nt * KT + kt)) * 32 + bl) * 16 + j] = (_Float16)w;
  }
}

// ---------------------------------------------------------------------------
// P2: X f32->f16 pre-swizzled to the WMMA 16-bit A 16x32 per-lane layout:
// lane row = lane&15; halfs[0..7] = K base..base+7, halfs[8..15] = base+16..,
// base = 8*(lane>>4).  Each thread: 8 contiguous f32 loads, one 16B store.
// ---------------------------------------------------------------------------
__global__ __launch_bounds__(256) void eml_prep_x(
    const float* __restrict__ X, _Float16* __restrict__ Aswz) {
  const int t    = blockIdx.x * blockDim.x + threadIdx.x; // [mt][kt][lane][h]
  const int h    = t & 1;
  const int lane = (t >> 1) & 31;
  const int kt   = (t >> 6) & (KT - 1);
  const int mt   = t >> 10;

  const int m     = mt * 16 + (lane & 15);
  const int kbase = kt * 32 + ((lane >> 4) * 8) + h * 16;

  const float* src = X + (size_t)m * NF + kbase;
  _Float16* dst = Aswz + ((((size_t)mt * KT + kt) * 32 + lane) * 16) + h * 8;
  #pragma unroll
  for (int j = 0; j < 8; ++j) dst[j] = (_Float16)src[j];
}

// ---------------------------------------------------------------------------
// GEMM + fused exp/leaf-sum epilogue.
// Grid: (BATCH/BR, NL/BC), 256 threads = 8 waves.
// ---------------------------------------------------------------------------
__global__ __launch_bounds__(256) void eml_gemm(
    const _Float16* __restrict__ Aswz, const _Float16* __restrict__ Bswz,
    const float* __restrict__ bias, float* __restrict__ partial) {
  __shared__ union {
    uint4 a4[BR / 16 * KT * 32 * 2];   // 8 m-tiles * 16 kt * 32 lanes * 32B = 128KB
    float red[BR][4];                  // reused after GEMM
  } sm;

  const int tid  = threadIdx.x;
  const int lane = tid & 31;
  const int wave = tid >> 5;
  const int wm   = wave & 1;     // 0..1 : which 64-row half
  const int wn   = wave >> 1;    // 0..3 : which 32-col slice

  // ---- stage A tile (contiguous 128KB, pre-swizzled) into LDS ----
  const unsigned long long gbase =
      (unsigned long long)(Aswz + (size_t)blockIdx.x * (BR / 16 * KT * 32 * 16));
  const unsigned ldsbase = (unsigned)(unsigned long long)&sm.a4[0];

#if __has_builtin(__builtin_amdgcn_tensor_load_to_lds)
  // Tensor Data Mover: one 1-line descriptor, data_size=8B, 16384 elements.
  if (wave == 0) {
    const unsigned n8 = (BR * NF * 2) / 8;   // 16384 8-byte units
    u32x4 g0;
    g0[0] = 1u;                               // count=1, is_restore=0, gather=0
    g0[1] = ldsbase;                          // lds_addr
    g0[2] = (unsigned)(gbase & 0xffffffffull);          // global_addr[31:0]
    g0[3] = (unsigned)((gbase >> 32) & 0x1ffffffull)    // global_addr[56:32]
            | (2u << 30);                               // type=2 ("image")
    i32x8 g1;
    g1[0] = (int)(3u << 16);                  // workgroup_mask=0, data_size=3 (8B)
    g1[1] = (int)((n8 & 0xffffu) << 16);      // tensor_dim0[15:0] @ bits 63:48
    g1[2] = (int)((n8 >> 16) | (1u << 16));   // tensor_dim0[31:16], tensor_dim1=1
    g1[3] = (int)(n8 << 16);                  // tile_dim0 = 16384 @ bits 127:112
    g1[4] = 1;                                // tile_dim1=1, tile_dim2=0
    g1[5] = (int)n8;                          // tensor_dim0_stride[31:0]
    g1[6] = 0;                                // stride high bits / dim1_stride lo
    g1[7] = 0;
    i32x4 gz = {0, 0, 0, 0};
#if __clang_major__ >= 23
    i32x8 gz8 = {0, 0, 0, 0, 0, 0, 0, 0};
    __builtin_amdgcn_tensor_load_to_lds(g0, g1, gz, gz, gz8, 0);
#else
    __builtin_amdgcn_tensor_load_to_lds(g0, g1, gz, gz, 0);
#endif
    __builtin_amdgcn_s_wait_tensorcnt(0);
  }
#else
  // Fallback: per-lane async global->LDS copies (ASYNCcnt tracked).
  #pragma unroll
  for (int i = 0; i < (BR * NF * 2) / (256 * 16); ++i) {  // 32 iters
    const unsigned l = ldsbase + (unsigned)(tid + i * 256) * 16u;
    const unsigned long long g = gbase + (unsigned long long)(tid + i * 256) * 16ull;
    asm volatile("global_load_async_to_lds_b128 %0, %1, off"
                 :: "v"(l), "v"(g) : "memory");
  }
  asm volatile("s_wait_asynccnt 0" ::: "memory");
#endif
  __syncthreads();

  v8f c[4][2];
  #pragma unroll
  for (int mi = 0; mi < 4; ++mi)
    #pragma unroll
    for (int ni = 0; ni < 2; ++ni) c[mi][ni] = (v8f){};

  const int ntg0 = blockIdx.y * (BC / 16) + wn * 2;
  const _Float16* bptr0 = Bswz + (size_t)(ntg0 + 0) * KT * 32 * 16;
  const _Float16* bptr1 = Bswz + (size_t)(ntg0 + 1) * KT * 32 * 16;

  union HF { uint4 u[2]; v16h h; };

  #pragma unroll 4     // bounded unroll: keep VGPRs < 256, 4-deep prefetch
  for (int kt = 0; kt < KT; ++kt) {
    HF a[4];
    #pragma unroll
    for (int mi = 0; mi < 4; ++mi) {
      const int base = (((wm * 4 + mi) * KT + kt) * 32 + lane) * 2;
      a[mi].u[0] = sm.a4[base + 0];      // ds_load_b128
      a[mi].u[1] = sm.a4[base + 1];      // ds_load_b128
    }
    v16h b0 = *(const v16h*)(bptr0 + ((size_t)kt * 32 + lane) * 16);
    v16h b1 = *(const v16h*)(bptr1 + ((size_t)kt * 32 + lane) * 16);
    #pragma unroll
    for (int mi = 0; mi < 4; ++mi) {
      c[mi][0] = __builtin_amdgcn_wmma_f32_16x16x32_f16(
          false, a[mi].h, false, b0, (short)0, c[mi][0], false, false);
      c[mi][1] = __builtin_amdgcn_wmma_f32_16x16x32_f16(
          false, a[mi].h, false, b1, (short)0, c[mi][1], false, false);
    }
  }

  // ---- epilogue: add bias, exp, sum this wave's 32 columns per row ----
  // C frag: VGPR v, lanes 0-15 -> (row v, col lane); lanes 16-31 -> (v+8, lane-16)
  const int colbase = blockIdx.y * BC + wn * 32;
  const float bias0 = bias[colbase + (lane & 15)];
  const float bias1 = bias[colbase + 16 + (lane & 15)];

  float s[4][8];
  #pragma unroll
  for (int mi = 0; mi < 4; ++mi)
    #pragma unroll
    for (int v = 0; v < 8; ++v)
      s[mi][v] = __expf(c[mi][0][v] + bias0) + __expf(c[mi][1][v] + bias1);

  #pragma unroll
  for (int m = 8; m >= 1; m >>= 1)
    #pragma unroll
    for (int mi = 0; mi < 4; ++mi)
      #pragma unroll
      for (int v = 0; v < 8; ++v)
        s[mi][v] += __shfl_xor(s[mi][v], m, 16);

  __syncthreads();               // LDS A done; reuse as sm.red
  if ((lane & 15) == 0) {
    const int hi = lane >> 4;
    #pragma unroll
    for (int mi = 0; mi < 4; ++mi)
      #pragma unroll
      for (int v = 0; v < 8; ++v)
        sm.red[wm * 64 + mi * 16 + hi * 8 + v][wn] = s[mi][v];
  }
  __syncthreads();

  if (tid < BR) {
    float t = sm.red[tid][0] + sm.red[tid][1] + sm.red[tid][2] + sm.red[tid][3];
    partial[(size_t)blockIdx.y * BATCH + (size_t)blockIdx.x * BR + tid] = t;
  }
}

// ---------------------------------------------------------------------------
__global__ __launch_bounds__(256) void eml_finalize(
    const float* __restrict__ partial, float* __restrict__ out) {
  const int b = blockIdx.x * blockDim.x + threadIdx.x;
  float s = 0.0f;
  #pragma unroll
  for (int cb = 0; cb < CB; ++cb) s += partial[(size_t)cb * BATCH + b];
  out[b] = __logf(s);
}

// ---------------------------------------------------------------------------
extern "C" void kernel_launch(void* const* d_in, const int* in_sizes, int n_in,
                              void* d_out, int out_size, void* d_ws,
                              size_t ws_size, hipStream_t stream) {
  (void)in_sizes; (void)n_in; (void)out_size; (void)ws_size;
  const float* X  = (const float*)d_in[0];   // (8192, 512) f32
  const float* ll = (const float*)d_in[1];   // (1024, 513) f32
  float* out = (float*)d_out;                // (8192,) f32

  char* ws = (char*)d_ws;
  _Float16* Aswz = (_Float16*)ws;                                  // 8 MB
  _Float16* Bswz = (_Float16*)(ws + (size_t)BATCH * NF * 2);       // 1 MB
  float* bias    = (float*)(ws + (size_t)(BATCH + NL) * NF * 2);   // 4 KB
  float* partial = bias + NL;                                      // 256 KB

  eml_prep_weights<<<NL / 8, 256, 0, stream>>>(ll, Bswz, bias);
  eml_prep_x<<<(BATCH / 16) * KT * 64 / 256, 256, 0, stream>>>(X, Aswz);

  dim3 grid(BATCH / BR, NL / BC);
  eml_gemm<<<grid, 256, 0, stream>>>(Aswz, Bswz, bias, partial);

  eml_finalize<<<BATCH / 256, 256, 0, stream>>>(partial, out);
}
